// LSSViewTransformer_23957327577708
// MI455X (gfx1250) — matrix operand order repaired and checked by reading, not compiled, not run on hardware
//
#include <hip/hip_runtime.h>
#include <hip/hip_bf16.h>
#include <math.h>

typedef __attribute__((ext_vector_type(16))) _Float16 v16h;
typedef __attribute__((ext_vector_type(8)))  float    v8f;

#define C_CH   64
#define D_BINS 48
#define IMG_N  12
#define FHH    16
#define FWW    44
#define HWIN   (FHH*FWW)          // 704
#define NPOS   (IMG_N*HWIN)       // 8448
#define BEV    200
#define NBEV   (BEV*BEV)          // 40000

// ---------------------------------------------------------------------------
// small utility kernels
// ---------------------------------------------------------------------------
__global__ void cvt_f32_to_f16(const float* __restrict__ src, _Float16* __restrict__ dst, int n) {
    int t = blockIdx.x * 256 + threadIdx.x;
    if (t < n) dst[t] = (_Float16)src[t];
}

__global__ void zero_f32(float* __restrict__ p, int n) {
    int t = blockIdx.x * 256 + threadIdx.x;
    if (t < n) p[t] = 0.0f;
}

// fused BN (+conv bias) parameters: out[set*128 + c] = scale, out[set*128+64+c] = bias
__global__ void fuse_bn(const float* cb0, const float* s0, const float* b0, const float* m0, const float* v0,
                        const float* cb1, const float* s1, const float* b1, const float* m1, const float* v1,
                        const float* cb2, const float* s2, const float* b2, const float* m2, const float* v2,
                        const float* cb3, const float* s3, const float* b3, const float* m3, const float* v3,
                        float* __restrict__ out) {
    int t = threadIdx.x;
    if (t >= 256) return;
    int set = t >> 6, c = t & 63;
    const float *cb, *s, *bb, *m, *v;
    if      (set == 0) { cb = cb0; s = s0; bb = b0; m = m0; v = v0; }
    else if (set == 1) { cb = cb1; s = s1; bb = b1; m = m1; v = v1; }
    else if (set == 2) { cb = cb2; s = s2; bb = b2; m = m2; v = v2; }
    else               { cb = cb3; s = s3; bb = b3; m = m3; v = v3; }
    float inv = s[c] * rsqrtf(v[c] + 1e-5f);
    out[set*128 + c]      = inv;
    out[set*128 + 64 + c] = (cb[c] - m[c]) * inv + bb[c];
}

// ---------------------------------------------------------------------------
// A-fragment weight packer.  Contiguous-k operand order used with V_WMMA_F32_16X16X32_F16
// A-matrix (16x32 f16): lane L holds row m = L&15; K-offset = (L<16)?0:8;
// element e (v = e>>1, half = e&1):
//   Klocal = 2*koff + e  (contiguous: k = 16*(L>>4) + e)
// dst index == flat thread id t = ((g*NKB + kb)*32 + lane)*16 + e
// mode 0: conv3x3, k = tap*64 + c, src = w[(m*64+c)*9 + tap]
// mode 1: direct,  src = w[m*K + k]
// ---------------------------------------------------------------------------
__global__ void pack_a(const float* __restrict__ w, _Float16* __restrict__ dst,
                       int G, int NKB, int K, int mode) {
    int t = blockIdx.x * 256 + threadIdx.x;
    int total = G * NKB * 512;
    if (t >= total) return;
    int e    = t & 15;
    int lane = (t >> 4) & 31;
    int kb   = (t >> 9) % NKB;
    int g    = t / (NKB * 512);
    int m    = g * 16 + (lane & 15);
    int koff = (lane & 16) ? 8 : 0;
    int v = e >> 1, half = e & 1;
    int kl = 2*koff + e;
    int k  = kb * 32 + kl;
    float val;
    if (mode == 0) { int tap = k >> 6; int c = k & 63; val = w[(m*64 + c)*9 + tap]; }
    else           { val = w[(size_t)m * K + k]; }
    dst[t] = (_Float16)val;
}

// ---------------------------------------------------------------------------
// implicit-GEMM 3x3 conv, C=64 -> C=64, fused scale/bias + ReLU.
// 4 waves; each wave computes a 16(Mchans) x 32(cols) tile via two
// independent accumulator chains (hides WMMA->WMMA latency, reuses each
// A-fragment for 2 WMMAs).  K = 9*64 = 576.
// LDS patch is channel-innermost: xs[(r*34+col)*64 + c]; a lane's 16
// B-fragment f16 values are one contiguous 32B LDS load.
// grid = imgs * H * ntiles, tile width 32.
// ---------------------------------------------------------------------------
template<bool F32OUT>
__global__ __launch_bounds__(128)
void conv3x3_wmma(const _Float16* __restrict__ in,
                  const _Float16* __restrict__ wpack,
                  const float* __restrict__ fscale,
                  const float* __restrict__ fbias,
                  _Float16* __restrict__ out_h,
                  float* __restrict__ out_f,
                  int H, int W, int ntiles) {
    __shared__ _Float16 xs[3 * 34 * 64];        // [(row r)*34 + col][c]
    const int HW = H * W;
    int bid = blockIdx.x;
    int wt  = bid % ntiles;
    int h   = (bid / ntiles) % H;
    int img = bid / (ntiles * H);
    int w0  = wt * 32;

    for (int t = threadIdx.x; t < 64 * 102; t += 128) {
        int c = t / 102, rem = t - c * 102;
        int r = rem / 34, col = rem - r * 34;
        int gh = h + r - 1, gw = w0 + col - 1;
        _Float16 v = (_Float16)0.f;
        if (gh >= 0 && gh < H && gw >= 0 && gw < W)
            v = in[((size_t)img * 64 + c) * HW + (size_t)gh * W + gw];
        xs[(r * 34 + col) * 64 + c] = v;
    }
    __syncthreads();

    int lane = threadIdx.x & 31;
    int g    = threadIdx.x >> 5;
    int n    = lane & 15;
    int kofs = lane & 16;

    v8f acc0 = {}, acc1 = {};
    for (int kb = 0; kb < 18; ++kb) {
        v16h a = *(const v16h*)(wpack + (((size_t)g * 18 + kb) * 32 + lane) * 16);
        int tap = kb >> 1;
        int kh = tap / 3, kw = tap - kh * 3;
        int c0 = (kb & 1) << 5;
        const _Float16* base = xs + c0 + kofs;
        v16h b0 = *(const v16h*)(base + (kh * 34 + n + kw) * 64);
        v16h b1 = *(const v16h*)(base + (kh * 34 + n + 16 + kw) * 64);
        acc0 = __builtin_amdgcn_wmma_f32_16x16x32_f16(false, a, false, b0, (short)0, acc0, false, false);
        acc1 = __builtin_amdgcn_wmma_f32_16x16x32_f16(false, a, false, b1, (short)0, acc1, false, false);
    }

    int mbase = g * 16 + ((lane >> 4) << 3);
    float4 sA = *(const float4*)(fscale + mbase);
    float4 sB = *(const float4*)(fscale + mbase + 4);
    float4 bA = *(const float4*)(fbias + mbase);
    float4 bB = *(const float4*)(fbias + mbase + 4);
    float sc[8] = {sA.x, sA.y, sA.z, sA.w, sB.x, sB.y, sB.z, sB.w};
    float bi[8] = {bA.x, bA.y, bA.z, bA.w, bB.x, bB.y, bB.z, bB.w};

    int wout0 = w0 + n;
    int wout1 = w0 + n + 16;
    if (wout0 < W) {
#pragma unroll
        for (int r = 0; r < 8; ++r) {
            float v = fmaxf(acc0[r] * sc[r] + bi[r], 0.f);
            size_t o = ((size_t)img * 64 + mbase + r) * HW + (size_t)h * W + wout0;
            if (F32OUT) out_f[o] = v; else out_h[o] = (_Float16)v;
        }
    }
    if (wout1 < W) {
#pragma unroll
        for (int r = 0; r < 8; ++r) {
            float v = fmaxf(acc1[r] * sc[r] + bi[r], 0.f);
            size_t o = ((size_t)img * 64 + mbase + r) * HW + (size_t)h * W + wout1;
            if (F32OUT) out_f[o] = v; else out_h[o] = (_Float16)v;
        }
    }
}

// ---------------------------------------------------------------------------
// depth/context head: 1x1 conv 64->112 (WMMA, 7 waves), then softmax over
// first 48 channels, context = channels 48..111.  LDS tile is [n][c].
// ---------------------------------------------------------------------------
__global__ __launch_bounds__(224)
void head_softmax(const _Float16* __restrict__ h2,
                  const _Float16* __restrict__ wpack3,
                  const float* __restrict__ b3,
                  _Float16* __restrict__ depth,
                  _Float16* __restrict__ ctx) {
    __shared__ _Float16 xs[16 * 64];            // [n][c]
    __shared__ float    ob[112 * 16];
    int P0  = blockIdx.x * 16;
    int img = P0 / HWIN;
    int pp  = P0 - img * HWIN;

    for (int t = threadIdx.x; t < 1024; t += 224) {
        int c = t >> 4, n = t & 15;
        xs[n * 64 + c] = h2[((size_t)img * 64 + c) * HWIN + pp + n];
    }
    __syncthreads();

    int lane = threadIdx.x & 31;
    int g    = threadIdx.x >> 5;
    int n    = lane & 15;
    int kofs = lane & 16;
    v8f acc = {};
#pragma unroll
    for (int kb = 0; kb < 2; ++kb) {
        v16h a = *(const v16h*)(wpack3 + (((size_t)g * 2 + kb) * 32 + lane) * 16);
        v16h b = *(const v16h*)(xs + (n * 64 + kb * 32 + kofs));
        acc = __builtin_amdgcn_wmma_f32_16x16x32_f16(false, a, false, b, (short)0, acc, false, false);
    }
    int mbase = g * 16 + ((lane >> 4) << 3);
#pragma unroll
    for (int r = 0; r < 8; ++r) {
        int m = mbase + r;
        ob[m * 16 + n] = acc[r] + b3[m];
    }
    __syncthreads();

    if (threadIdx.x < 16) {
        int nn = threadIdx.x;
        float mx = -1e30f;
        for (int c = 0; c < 48; ++c) mx = fmaxf(mx, ob[c * 16 + nn]);
        float s = 0.f;
        for (int c = 0; c < 48; ++c) s += expf(ob[c * 16 + nn] - mx);
        float inv = 1.f / s;
        for (int c = 0; c < 48; ++c)
            depth[((size_t)img * 48 + c) * HWIN + pp + nn] = (_Float16)(expf(ob[c * 16 + nn] - mx) * inv);
    }
    for (int t = threadIdx.x; t < 1024; t += 224) {
        int c = t >> 4, nn = t & 15;
        ctx[((size_t)img * 64 + c) * HWIN + pp + nn] = (_Float16)ob[(48 + c) * 16 + nn];
    }
}

// ---------------------------------------------------------------------------
// fused lift + rd 1x1 conv:  out[m,p] = ReLU(BN( sum_{d,c} W[m,d*64+c] *
// depth[d,p] * ctx[c,p] )).  GEMM M=64, K=3072, N=32 per block; B built on
// the fly (contiguous 32B LDS load of context, broadcast f16 multiply by
// depth).  Two accumulator chains per wave reuse each A-fragment twice.
// feat stored position-major ([p][c]) for the scatter stage.
// ---------------------------------------------------------------------------
__global__ __launch_bounds__(128)
void lift_reduce(const _Float16* __restrict__ depth,
                 const _Float16* __restrict__ ctx,
                 const _Float16* __restrict__ wpackrd,
                 const float* __restrict__ fscale,
                 const float* __restrict__ fbias,
                 float* __restrict__ feat) {
    __shared__ _Float16 dep[48 * 32];           // [d][n32]
    __shared__ _Float16 cl [32 * 64];           // [n32][c]
    int P0  = blockIdx.x * 32;
    int img = P0 / HWIN;
    int pp  = P0 - img * HWIN;
    for (int t = threadIdx.x; t < 48 * 32; t += 128) {
        int d = t >> 5, n = t & 31;
        dep[t] = depth[((size_t)img * 48 + d) * HWIN + pp + n];
    }
    for (int t = threadIdx.x; t < 64 * 32; t += 128) {
        int c = t >> 5, n = t & 31;
        cl[n * 64 + c] = ctx[((size_t)img * 64 + c) * HWIN + pp + n];
    }
    __syncthreads();

    int lane = threadIdx.x & 31;
    int g    = threadIdx.x >> 5;
    int n    = lane & 15;
    int kofs = lane & 16;
    v8f acc0 = {}, acc1 = {};
    for (int kb = 0; kb < 96; ++kb) {
        v16h a = *(const v16h*)(wpackrd + (((size_t)g * 96 + kb) * 32 + lane) * 16);
        int d  = kb >> 1;
        int c0 = (kb & 1) << 5;
        _Float16 hd0 = dep[d * 32 + n];
        _Float16 hd1 = dep[d * 32 + n + 16];
        v16h cv0 = *(const v16h*)(cl + (n * 64 + c0 + kofs));
        v16h cv1 = *(const v16h*)(cl + ((n + 16) * 64 + c0 + kofs));
        v16h b0, b1;
#pragma unroll
        for (int e = 0; e < 16; ++e) { b0[e] = hd0 * cv0[e]; b1[e] = hd1 * cv1[e]; }
        acc0 = __builtin_amdgcn_wmma_f32_16x16x32_f16(false, a, false, b0, (short)0, acc0, false, false);
        acc1 = __builtin_amdgcn_wmma_f32_16x16x32_f16(false, a, false, b1, (short)0, acc1, false, false);
    }
    int mbase = g * 16 + ((lane >> 4) << 3);
    float4 sA = *(const float4*)(fscale + mbase);
    float4 sB = *(const float4*)(fscale + mbase + 4);
    float4 bA = *(const float4*)(fbias + mbase);
    float4 bB = *(const float4*)(fbias + mbase + 4);
    float sc[8] = {sA.x, sA.y, sA.z, sA.w, sB.x, sB.y, sB.z, sB.w};
    float bi[8] = {bA.x, bA.y, bA.z, bA.w, bB.x, bB.y, bB.z, bB.w};
    float* fp0 = feat + ((size_t)(P0 + n)) * 64 + mbase;
    float* fp1 = feat + ((size_t)(P0 + n + 16)) * 64 + mbase;
#pragma unroll
    for (int r = 0; r < 8; ++r) {
        fp0[r] = fmaxf(acc0[r] * sc[r] + bi[r], 0.f);
        fp1[r] = fmaxf(acc1[r] * sc[r] + bi[r], 0.f);
    }
}

// ---------------------------------------------------------------------------
// geometry + scatter-mean.  One wave per (b,n,h,w) ray; lanes carry channels.
// Consecutive depth bins mapping to the same BEV cell are merged (multiplicity)
// before issuing f32 global atomics.
// ---------------------------------------------------------------------------
__global__ __launch_bounds__(256)
void scatter_bev(const float* __restrict__ feat,
                 const float* __restrict__ intr,
                 const float* __restrict__ extr,
                 float* __restrict__ bev_sum,
                 float* __restrict__ cnt) {
    int wid  = (blockIdx.x * 256 + threadIdx.x) >> 5;
    int lane = threadIdx.x & 31;
    if (wid >= NPOS) return;
    int img = wid / HWIN;
    int pp  = wid - img * HWIN;
    int h = pp / FWW, w = pp - h * FWW;
    int b = img / 6;
    const float* Km = intr + (size_t)img * 9;
    const float* E  = extr + (size_t)img * 16;
    float fx = Km[0], cx = Km[2], fy = Km[4], cy = Km[5];
    float e00 = E[0], e01 = E[1], e02 = E[2], e03 = E[3];
    float e10 = E[4], e11 = E[5], e12 = E[6], e13 = E[7];
    float ax = ((float)w * 16.0f - cx) / fx;   // IMG_W/FW = 704/44 = 16
    float ay = ((float)h * 16.0f - cy) / fy;   // IMG_H/FH = 256/16 = 16
    const float* fb = feat + (size_t)wid * 64;
    float f0 = fb[lane], f1 = fb[lane + 32];
    float* bs = bev_sum + (size_t)b * NBEV * 64;
    float* cb = cnt + (size_t)b * NBEV;

    int prev = -1; float mult = 0.f;
    for (int di = 0; di < 48; ++di) {
        float dv = 2.0f + (56.0f / 47.0f) * (float)di;
        float Xc = ax * dv, Yc = ay * dv;
        float Xe = e00 * Xc + e01 * Yc + e02 * dv + e03;
        float Ye = e10 * Xc + e11 * Yc + e12 * dv + e13;
        int col = (int)((Xe + 50.f) * 2.f);    // truncation matches astype(int32)
        int row = (int)((Ye + 50.f) * 2.f);
        int idx = (col >= 0 && col < BEV && row >= 0 && row < BEV) ? row * BEV + col : -1;
        if (idx == prev) { mult += 1.f; }
        else {
            if (prev >= 0) {
                atomicAdd(bs + (size_t)prev * 64 + lane,      mult * f0);
                atomicAdd(bs + (size_t)prev * 64 + lane + 32, mult * f1);
                if (lane == 0) atomicAdd(cb + prev, mult);
            }
            prev = idx; mult = 1.f;
        }
    }
    if (prev >= 0) {
        atomicAdd(bs + (size_t)prev * 64 + lane,      mult * f0);
        atomicAdd(bs + (size_t)prev * 64 + lane + 32, mult * f1);
        if (lane == 0) atomicAdd(cb + prev, mult);
    }
}

// normalize bev_sum ([b][pix][c]) by counts -> f16 NCHW for the final conv
__global__ void bev_norm(const float* __restrict__ bev_sum, const float* __restrict__ cnt,
                         _Float16* __restrict__ bev_h) {
    int t = blockIdx.x * 256 + threadIdx.x;
    if (t >= 2 * 64 * NBEV) return;
    int b = t / (64 * NBEV);
    int rem = t - b * 64 * NBEV;
    int c = rem / NBEV;
    int pix = rem - c * NBEV;
    float s = bev_sum[((size_t)b * NBEV + pix) * 64 + c];
    float n = cnt[(size_t)b * NBEV + pix];
    bev_h[t] = (_Float16)(s / fmaxf(n, 1.0f));
}

// ---------------------------------------------------------------------------
extern "C" void kernel_launch(void* const* d_in, const int* in_sizes, int n_in,
                              void* d_out, int out_size, void* d_ws, size_t ws_size,
                              hipStream_t stream) {
    if (n_in < 29) return;
    const float* feats = (const float*)d_in[0];
    const float* intr  = (const float*)d_in[1];
    const float* extr  = (const float*)d_in[2];
    const float* dn_w1 = (const float*)d_in[3];
    const float* dn_b1 = (const float*)d_in[4];
    const float* bn1s  = (const float*)d_in[5];
    const float* bn1b  = (const float*)d_in[6];
    const float* bn1m  = (const float*)d_in[7];
    const float* bn1v  = (const float*)d_in[8];
    const float* dn_w2 = (const float*)d_in[9];
    const float* dn_b2 = (const float*)d_in[10];
    const float* bn2s  = (const float*)d_in[11];
    const float* bn2b  = (const float*)d_in[12];
    const float* bn2m  = (const float*)d_in[13];
    const float* bn2v  = (const float*)d_in[14];
    const float* dn_w3 = (const float*)d_in[15];
    const float* dn_b3 = (const float*)d_in[16];
    const float* rd_w  = (const float*)d_in[17];
    const float* rd_b  = (const float*)d_in[18];
    const float* rbns  = (const float*)d_in[19];
    const float* rbnb  = (const float*)d_in[20];
    const float* rbnm  = (const float*)d_in[21];
    const float* rbnv  = (const float*)d_in[22];
    const float* ag_w  = (const float*)d_in[23];
    const float* ag_b  = (const float*)d_in[24];
    const float* abns  = (const float*)d_in[25];
    const float* abnb  = (const float*)d_in[26];
    const float* abnm  = (const float*)d_in[27];
    const float* abnv  = (const float*)d_in[28];

    // workspace carve-up (256B aligned)
    size_t off = 0;
    char* ws = (char*)d_ws;
    auto carve = [&](size_t bytes) -> char* {
        char* p = ws + off;
        off = (off + bytes + 255) & ~(size_t)255;
        return p;
    };
    _Float16* h0     = (_Float16*)carve((size_t)NPOS * 64 * 2);
    _Float16* h1     = (_Float16*)carve((size_t)NPOS * 64 * 2);
    _Float16* h2     = (_Float16*)carve((size_t)NPOS * 64 * 2);
    _Float16* depth  = (_Float16*)carve((size_t)NPOS * 48 * 2);
    _Float16* ctx    = (_Float16*)carve((size_t)NPOS * 64 * 2);
    float*    feat   = (float*)   carve((size_t)NPOS * 64 * 4);
    float*    bevsum = (float*)   carve((size_t)2 * NBEV * 64 * 4);
    float*    cntb   = (float*)   carve((size_t)2 * NBEV * 4);
    _Float16* bev_h  = (_Float16*)carve((size_t)2 * NBEV * 64 * 2);
    _Float16* wp1    = (_Float16*)carve((size_t)4 * 18 * 512 * 2);
    _Float16* wp2    = (_Float16*)carve((size_t)4 * 18 * 512 * 2);
    _Float16* wpag   = (_Float16*)carve((size_t)4 * 18 * 512 * 2);
    _Float16* wp3    = (_Float16*)carve((size_t)7 * 2  * 512 * 2);
    _Float16* wprd   = (_Float16*)carve((size_t)4 * 96 * 512 * 2);
    float*    fused  = (float*)   carve(4 * 128 * 4);
    (void)ws_size; (void)in_sizes; (void)out_size;

    // 1. feats f32 -> f16
    cvt_f32_to_f16<<<(NPOS * 64 + 255) / 256, 256, 0, stream>>>(feats, h0, NPOS * 64);

    // 2. fused BN params (dn1, dn2, rd, ag)
    fuse_bn<<<1, 256, 0, stream>>>(dn_b1, bn1s, bn1b, bn1m, bn1v,
                                   dn_b2, bn2s, bn2b, bn2m, bn2v,
                                   rd_b,  rbns, rbnb, rbnm, rbnv,
                                   ag_b,  abns, abnb, abnm, abnv, fused);

    // 3. pack weights into WMMA A-fragment order
    pack_a<<<(4 * 18 * 512 + 255) / 256, 256, 0, stream>>>(dn_w1, wp1,  4, 18, 576,  0);
    pack_a<<<(4 * 18 * 512 + 255) / 256, 256, 0, stream>>>(dn_w2, wp2,  4, 18, 576,  0);
    pack_a<<<(4 * 18 * 512 + 255) / 256, 256, 0, stream>>>(ag_w,  wpag, 4, 18, 576,  0);
    pack_a<<<(7 * 2  * 512 + 255) / 256, 256, 0, stream>>>(dn_w3, wp3,  7, 2,  64,   1);
    pack_a<<<(4 * 96 * 512 + 255) / 256, 256, 0, stream>>>(rd_w,  wprd, 4, 96, 3072, 1);

    // 4. conv1 + conv2 (implicit GEMM, WMMA, 32-wide tiles)
    conv3x3_wmma<false><<<IMG_N * FHH * 2, 128, 0, stream>>>(h0, wp1, fused + 0,   fused + 64,  h1, nullptr, FHH, FWW, 2);
    conv3x3_wmma<false><<<IMG_N * FHH * 2, 128, 0, stream>>>(h1, wp2, fused + 128, fused + 192, h2, nullptr, FHH, FWW, 2);

    // 5. depth/context head + softmax
    head_softmax<<<NPOS / 16, 224, 0, stream>>>(h2, wp3, dn_b3, depth, ctx);

    // 6. fused lift + rd reduction (K=3072 WMMA GEMM, B built on the fly)
    lift_reduce<<<NPOS / 32, 128, 0, stream>>>(depth, ctx, wprd, fused + 256, fused + 320, feat);

    // 7. zero BEV accumulators, scatter-mean
    zero_f32<<<(2 * NBEV * 64 + 255) / 256, 256, 0, stream>>>(bevsum, 2 * NBEV * 64);
    zero_f32<<<(2 * NBEV + 255) / 256, 256, 0, stream>>>(cntb, 2 * NBEV);
    scatter_bev<<<(NPOS * 32 + 255) / 256, 256, 0, stream>>>(feat, intr, extr, bevsum, cntb);

    // 8. normalize -> f16 NCHW
    bev_norm<<<(2 * 64 * NBEV + 255) / 256, 256, 0, stream>>>(bevsum, cntb, bev_h);

    // 9. final BEV 3x3 conv (WMMA, 32-wide tiles) -> f32 output
    conv3x3_wmma<true><<<2 * BEV * 7, 128, 0, stream>>>(bev_h, wpag, fused + 384, fused + 448,
                                                        nullptr, (float*)d_out, BEV, BEV, 7);
}
